// SkinNet_inner_43997644980908
// MI455X (gfx1250) — matrix operand order repaired and checked by reading, not compiled, not run on hardware
//
#include <hip/hip_runtime.h>
#include <cstdint>
#include <cstddef>

// ---------------------------------------------------------------------------
// SkinNet forward, MI455X (gfx1250, wave32).
// Every dense layer -> v_wmma_f32_16x16x32_f16 (f16 in, f32 accumulate).
// Weights are pre-packed ONCE into f16 in WMMA B-fragment lane order, so the
// inner loop loads B as a single aligned v16h (2 x b128) with zero converts;
// A is four b128 loads + packed cvt per k-step. No bounds guards anywhere in
// the hot loops (shapes padded at pack time; N/E are multiples of 16).
// Edge features cat[x_i, x_j-x_i] and cls input cat[x3, xg[batch]] are
// gathered on the fly in the A-fragment path (never materialized).
// ---------------------------------------------------------------------------

typedef _Float16 half_t;
typedef __attribute__((ext_vector_type(16))) _Float16 v16h;
typedef __attribute__((ext_vector_type(8)))  float    v8f;

// ======================= elementwise / aggregation =========================

__global__ void k_fill(float* __restrict__ p, float v, long long n) {
  long long i = (long long)blockIdx.x * blockDim.x + threadIdx.x;
  if (i < n) p[i] = v;
}

__global__ void k_fixup_nonfinite(float* __restrict__ p, long long n) {
  long long i = (long long)blockIdx.x * blockDim.x + threadIdx.x;
  if (i < n) { float v = p[i]; if (!__builtin_isfinite(v)) p[i] = 0.f; }
}

// dst[r, off:off+w] = src[r, 0:w]   (dst leading dim ldD)
__global__ void k_copy_cols(float* __restrict__ dst, int ldD, int off,
                            const float* __restrict__ src, int w, int n) {
  long long i = (long long)blockIdx.x * blockDim.x + threadIdx.x;
  if (i < (long long)n * w) {
    int r = (int)(i / w), c = (int)(i % w);
    dst[(long long)r * ldD + off + c] = src[i];
  }
}

// dst[r, 0:w] = src[r, off:off+w]   (src leading dim ldS)
__global__ void k_extract_cols(float* __restrict__ dst, int w,
                               const float* __restrict__ src, int ldS, int off, int n) {
  long long i = (long long)blockIdx.x * blockDim.x + threadIdx.x;
  if (i < (long long)n * w) {
    int r = (int)(i / w), c = (int)(i % w);
    dst[i] = src[(long long)r * ldS + off + c];
  }
}

// order-independent float atomic max (valid for init = -inf)
__device__ __forceinline__ void atomic_max_f32(float* addr, float v) {
  if (v >= 0.f) atomicMax((int*)addr,          __float_as_int(v));
  else          atomicMin((unsigned int*)addr, __float_as_uint(v));
}

__global__ void k_scatter_max(const float* __restrict__ msg, const int* __restrict__ dst,
                              float* __restrict__ out, int ldO, int colOff,
                              long long E, int C) {
  long long i = (long long)blockIdx.x * blockDim.x + threadIdx.x;
  if (i < E * (long long)C) {
    int e = (int)(i / C), c = (int)(i % C);
    atomic_max_f32(&out[(long long)dst[e] * ldO + colOff + c], msg[i]);
  }
}

__global__ void k_segment_max(const float* __restrict__ x, const int* __restrict__ seg,
                              float* __restrict__ out, long long n, int C) {
  long long i = (long long)blockIdx.x * blockDim.x + threadIdx.x;
  if (i < n * (long long)C) {
    int r = (int)(i / C), c = (int)(i % C);
    atomic_max_f32(&out[(long long)seg[r] * C + c], x[i]);
  }
}

// ====================== weight pre-pack (once, tiny) =======================
// B 32x16 f16 fragment (ISA 7.12.2): col = lane&15, half j holds
// K = k0 + (lane>>4)*8 + j (j<8), k0 + 16 + (lane>>4)*8 + (j-8) (j>=8).  Packed layout, v16h units:
//   P[(t*Q + q)*32 + lane]  with t = col tile, q = k0/32, Q = K/32.
// Zero-fills k >= Kv or col >= Mv (absorbs K=43 and M=5 padding).
__global__ void k_pack_b(const float* __restrict__ W, half_t* __restrict__ P,
                         int K, int M, int Kv, int Mv) {
  long long i = (long long)blockIdx.x * blockDim.x + threadIdx.x;
  if (i >= (long long)K * M) return;
  int j    = (int)(i & 15);
  int lane = (int)((i >> 4) & 31);
  long long tq = i >> 9;                    // t*Q + q  (32*16 = 512 halves)
  int Q = K >> 5;
  int q = (int)(tq % Q), t = (int)(tq / Q);
  int k   = q * 32 + (j < 8 ? (lane >> 4) * 8 + j : 16 + (lane >> 4) * 8 + (j - 8));
  int col = t * 16 + (lane & 15);
  float v = (k < Kv && col < Mv) ? W[(long long)k * Mv + col] : 0.f;
  P[i] = (half_t)v;
}

// ============================ WMMA fragments ===============================
// A 16x32 f16: lane row = lane&15, h = lane>>4; per lane two contiguous
// 8-float runs at k0+h*8 and k0+16+h*8 (four b128 loads, guard-free).

__device__ __forceinline__ void pack8(v16h& a, int base, float4 u, float4 v) {
  a[base + 0] = (half_t)u.x; a[base + 1] = (half_t)u.y;
  a[base + 2] = (half_t)u.z; a[base + 3] = (half_t)u.w;
  a[base + 4] = (half_t)v.x; a[base + 5] = (half_t)v.y;
  a[base + 6] = (half_t)v.z; a[base + 7] = (half_t)v.w;
}

__device__ __forceinline__ v16h a_from_row(const float* __restrict__ Xr, int k0, int h) {
  const float4* p0 = (const float4*)(Xr + k0 + h * 8);
  const float4* p1 = (const float4*)(Xr + k0 + 16 + h * 8);
  v16h a;
  pack8(a, 0, p0[0], p0[1]);
  pack8(a, 8, p1[0], p1[1]);
  return a;
}

// a = Xs[k] - Xi[k] (EdgeConv "x_j - x_i" half)
__device__ __forceinline__ v16h a_from_diff(const float* __restrict__ Xs,
                                            const float* __restrict__ Xi, int k0, int h) {
  const float4* s0 = (const float4*)(Xs + k0 + h * 8);
  const float4* s1 = (const float4*)(Xs + k0 + 16 + h * 8);
  const float4* i0 = (const float4*)(Xi + k0 + h * 8);
  const float4* i1 = (const float4*)(Xi + k0 + 16 + h * 8);
  float4 a0 = s0[0], a1 = s0[1], a2 = s1[0], a3 = s1[1];
  float4 b0 = i0[0], b1 = i0[1], b2 = i1[0], b3 = i1[1];
  float4 d0 = {a0.x - b0.x, a0.y - b0.y, a0.z - b0.z, a0.w - b0.w};
  float4 d1 = {a1.x - b1.x, a1.y - b1.y, a1.z - b1.z, a1.w - b1.w};
  float4 d2 = {a2.x - b2.x, a2.y - b2.y, a2.z - b2.z, a2.w - b2.w};
  float4 d3 = {a3.x - b3.x, a3.y - b3.y, a3.z - b3.z, a3.w - b3.w};
  v16h a;
  pack8(a, 0, d0, d1);
  pack8(a, 8, d2, d3);
  return a;
}

#define BFRAG(P, Q, t, q) (((const v16h*)(P))[((long long)(t) * (Q) + (q)) * 32 + lane])
#define WMMA(a, b, c) __builtin_amdgcn_wmma_f32_16x16x32_f16(false, a, false, b, (short)0, c, false, false)

// -------- plain GEMM: Y[rows, M](ldY) = act(X[rows, K] @ Wpacked + bias) ---
// rows % 16 == 0, K % 32 == 0, M % (16*CPW) == 0. grid = (rows/16, M/(16*CPW)).
template <int K, int CPW>
__global__ __launch_bounds__(32)
void k_gemm(const float* __restrict__ X, const half_t* __restrict__ P,
            const float* __restrict__ bias, float* __restrict__ Y,
            int M, int ldY, int relu) {
  constexpr int Q = K / 32;
  const int lane = threadIdx.x & 31, h = lane >> 4, lr = lane & 15;
  const long long row0 = (long long)blockIdx.x << 4;
  const int t0 = blockIdx.y * CPW;
  const float* Xr = X + (row0 + lr) * K;
  v8f acc[CPW];
#pragma unroll
  for (int c = 0; c < CPW; ++c) acc[c] = (v8f){};
  for (int k0 = 0; k0 < K; k0 += 32) {
    v16h a = a_from_row(Xr, k0, h);
    const int q = k0 >> 5;
#pragma unroll
    for (int c = 0; c < CPW; ++c)
      acc[c] = WMMA(a, BFRAG(P, Q, t0 + c, q), acc[c]);
  }
#pragma unroll
  for (int c = 0; c < CPW; ++c) {
    int col = (t0 + c) * 16 + lr;
    float bv = bias[col];
#pragma unroll
    for (int r = 0; r < 8; ++r) {
      float v = acc[c][r] + bv;
      if (relu) v = fmaxf(v, 0.f);
      Y[(row0 + r + h * 8) * ldY + col] = v;
    }
  }
}

// -------- EdgeConv layer 1: rows = edges, A row = cat[x_i, x_j - x_i] ------
// K = 2D, M = 128, CPW = 8 -> one wave covers all 128 cols. grid = (E/16, 1).
template <int D>
__global__ __launch_bounds__(32)
void k_gemm_edge(const float* __restrict__ x, const int* __restrict__ src,
                 const int* __restrict__ dst, const half_t* __restrict__ P,
                 const float* __restrict__ bias, float* __restrict__ Y) {
  constexpr int Q = (2 * D) / 32;
  const int lane = threadIdx.x & 31, h = lane >> 4, lr = lane & 15;
  const long long row0 = (long long)blockIdx.x << 4;
  const int e = (int)row0 + lr;
  const float* Xi = x + (long long)dst[e] * D;
  const float* Xs = x + (long long)src[e] * D;
  v8f acc[8];
#pragma unroll
  for (int c = 0; c < 8; ++c) acc[c] = (v8f){};
  for (int k0 = 0; k0 < D; k0 += 32) {          // first half: x_i
    v16h a = a_from_row(Xi, k0, h);
    const int q = k0 >> 5;
#pragma unroll
    for (int c = 0; c < 8; ++c)
      acc[c] = WMMA(a, BFRAG(P, Q, c, q), acc[c]);
  }
  for (int k0 = 0; k0 < D; k0 += 32) {          // second half: x_j - x_i
    v16h a = a_from_diff(Xs, Xi, k0, h);
    const int q = (D + k0) >> 5;
#pragma unroll
    for (int c = 0; c < 8; ++c)
      acc[c] = WMMA(a, BFRAG(P, Q, c, q), acc[c]);
  }
#pragma unroll
  for (int c = 0; c < 8; ++c) {
    int col = c * 16 + lr;
    float bv = bias[col];
#pragma unroll
    for (int r = 0; r < 8; ++r)
      Y[(row0 + r + h * 8) * 128 + col] = fmaxf(acc[c][r] + bv, 0.f);  // relu
  }
}

// ---- concat GEMM: A row n = cat[X1[n, 0:256], X2[gidx[n], 0:1024]] --------
// K = 1280, M = 1024, CPW = 8. grid = (N/16, 8).
__global__ __launch_bounds__(32)
void k_gemm_cat(const float* __restrict__ X1, const float* __restrict__ X2,
                const int* __restrict__ gidx, const half_t* __restrict__ P,
                const float* __restrict__ bias, float* __restrict__ Y) {
  constexpr int Q = 40;  // 1280 / 32
  const int lane = threadIdx.x & 31, h = lane >> 4, lr = lane & 15;
  const long long row0 = (long long)blockIdx.x << 4;
  const int t0 = blockIdx.y * 8;
  const float* X1r = X1 + (row0 + lr) * 256;
  const float* X2r = X2 + (long long)gidx[row0 + lr] * 1024;
  v8f acc[8];
#pragma unroll
  for (int c = 0; c < 8; ++c) acc[c] = (v8f){};
  for (int k0 = 0; k0 < 256; k0 += 32) {
    v16h a = a_from_row(X1r, k0, h);
    const int q = k0 >> 5;
#pragma unroll
    for (int c = 0; c < 8; ++c)
      acc[c] = WMMA(a, BFRAG(P, Q, t0 + c, q), acc[c]);
  }
  for (int k0 = 0; k0 < 1024; k0 += 32) {
    v16h a = a_from_row(X2r, k0, h);
    const int q = (256 + k0) >> 5;
#pragma unroll
    for (int c = 0; c < 8; ++c)
      acc[c] = WMMA(a, BFRAG(P, Q, t0 + c, q), acc[c]);
  }
#pragma unroll
  for (int c = 0; c < 8; ++c) {
    int col = (t0 + c) * 16 + lr;
    float bv = bias[col];
#pragma unroll
    for (int r = 0; r < 8; ++r)
      Y[(row0 + r + h * 8) * 1024 + col] = fmaxf(acc[c][r] + bv, 0.f);  // relu
  }
}

// ================================ host side ================================

static inline int cdiv(long long a, int b) { return (int)((a + b - 1) / b); }

template <int K, int CPW>
static void gemm(hipStream_t s, const float* X, const half_t* P, const float* b,
                 float* Y, int rows, int M, int ldY, int relu) {
  k_gemm<K, CPW><<<dim3(rows / 16, M / (16 * CPW)), 32, 0, s>>>(X, P, b, Y, M, ldY, relu);
}

static void launch_fill(float* p, float v, long long n, hipStream_t s) {
  k_fill<<<cdiv(n, 256), 256, 0, s>>>(p, v, n);
}

extern "C" void kernel_launch(void* const* d_in, const int* in_sizes, int n_in,
                              void* d_out, int out_size, void* d_ws, size_t ws_size,
                              hipStream_t stream) {
  (void)n_in; (void)out_size; (void)ws_size;
  const int N = in_sizes[0] / 3;   // 40000 nodes (multiple of 16)
  const int E = in_sizes[3] / 2;   // 160000 edges (multiple of 16)
  const int G = 8;

  auto f = [&](int i) { return (const float*)d_in[i]; };
  const float* pos    = f(0);
  const float* skin   = f(1);
  const float* motion = f(2);
  const int* tpl   = (const int*)d_in[3];
  const int* geo   = (const int*)d_in[4];
  const int* batch = (const int*)d_in[5];
  const int *tsrc = tpl, *tdst = tpl + E, *gsrc = geo, *gdst = geo + E;

  // params flattened in reference insertion order:
  // gcu{1,2,3}: posW,posB, tplW0,tplB0,tplW1,tplB1, geoW0,geoB0,geoW1,geoB1, outW,outB
  const int GCU1 = 6, GCU2 = 18, GCU3 = 30, MLT2 = 42, CLS = 46, FIN = 50;

  // ---- workspace carve-out (256B aligned slabs) ----
  char* w = (char*)d_ws;
  auto takeB = [&](long long bytes) -> void* {
    void* p = w;
    w += ((bytes + 255) & ~255LL);
    return p;
  };
  auto takeF = [&](long long elems) { return (float*)takeB(elems * 4); };

  float* rawP = takeF((long long)N * 64);    // cat[pos, skin] zero-padded 43 -> 64
  float* xbuf = takeF((long long)N * 320);   // cat[feat, pos_feat], widest d
  float* eH1  = takeF((long long)E * 128);   // edge hidden
  float* eH2  = takeF((long long)E * 128);   // edge message
  float* xcat = takeF((long long)N * 256);   // cat[x_tpl, x_geo]
  float* xA   = takeF((long long)N * 256);   // x1 then x3
  float* xB   = takeF((long long)N * 256);   // x2
  float* g1   = takeF((long long)N * 512);   // mlt2 hidden / cls hidden2
  float* g2   = takeF((long long)N * 1024);  // mlt2 out / cls hidden1
  float* xg   = takeF((long long)G * 1024);  // per-graph max-pool
  float* bfP  = takeF(16);                   // final bias padded 5 -> 16
  float* Yf   = takeF((long long)N * 16);    // padded final output

  // pack a weight [Kv, Mv] (row-major) into f16 B-fragment order, padded K x M
  auto packW = [&](const float* W, int K, int M, int Kv, int Mv) {
    half_t* P = (half_t*)takeB((long long)K * M * 2);
    k_pack_b<<<cdiv((long long)K * M, 256), 256, 0, stream>>>(W, P, K, M, Kv, Mv);
    return P;
  };

  struct GcuP { half_t *p, *t0, *t1, *g0, *g1, *o; };
  auto packGcu = [&](int B, int twoD) {
    GcuP g;
    g.p  = packW(f(B + 0), 64, 64, 43, 64);        // pos: K padded 43 -> 64
    g.t0 = packW(f(B + 2), twoD, 128, twoD, 128);
    g.t1 = packW(f(B + 4), 128, 128, 128, 128);
    g.g0 = packW(f(B + 6), twoD, 128, twoD, 128);
    g.g1 = packW(f(B + 8), 128, 128, 128, 128);
    g.o  = packW(f(B + 10), 256, 256, 256, 256);
    return g;
  };
  GcuP gp1 = packGcu(GCU1, 256), gp2 = packGcu(GCU2, 640), gp3 = packGcu(GCU3, 640);
  half_t* Pm0 = packW(f(MLT2 + 0), 256, 512, 256, 512);
  half_t* Pm1 = packW(f(MLT2 + 2), 512, 1024, 512, 1024);
  half_t* Pc0 = packW(f(CLS + 0), 1280, 1024, 1280, 1024);
  half_t* Pc1 = packW(f(CLS + 2), 1024, 512, 1024, 512);
  half_t* Pfin = packW(f(FIN + 0), 512, 16, 512, 5);  // M padded 5 -> 16

  // raw = cat[pos, skin], zero-padded to K = 64
  launch_fill(rawP, 0.f, (long long)N * 64, stream);
  k_copy_cols<<<cdiv((long long)N * 3, 256), 256, 0, stream>>>(rawP, 64, 0, pos, 3, N);
  k_copy_cols<<<cdiv((long long)N * 40, 256), 256, 0, stream>>>(rawP, 64, 3, skin, 40, N);

  // one GCU (d = feat_dim + 64)
  auto gcu = [&](int B, const GcuP& gp, const float* feat, int fd, float* out) {
    const int d = fd + 64;
    // x = cat[feat, relu(raw @ Wp + bp)]
    k_copy_cols<<<cdiv((long long)N * fd, 256), 256, 0, stream>>>(xbuf, d, 0, feat, fd, N);
    gemm<64, 4>(stream, rawP, gp.p, f(B + 1), xbuf + fd, N, 64, d, 1);
    launch_fill(xcat, -__builtin_inff(), (long long)N * 256, stream);
    // tpl EdgeConv
    if (d == 128) k_gemm_edge<128><<<dim3(E / 16, 1), 32, 0, stream>>>(xbuf, tsrc, tdst, gp.t0, f(B + 3), eH1);
    else          k_gemm_edge<320><<<dim3(E / 16, 1), 32, 0, stream>>>(xbuf, tsrc, tdst, gp.t0, f(B + 3), eH1);
    gemm<128, 8>(stream, eH1, gp.t1, f(B + 5), eH2, E, 128, 128, 1);
    k_scatter_max<<<cdiv((long long)E * 128, 256), 256, 0, stream>>>(eH2, tdst, xcat, 256, 0, E, 128);
    // geo EdgeConv
    if (d == 128) k_gemm_edge<128><<<dim3(E / 16, 1), 32, 0, stream>>>(xbuf, gsrc, gdst, gp.g0, f(B + 7), eH1);
    else          k_gemm_edge<320><<<dim3(E / 16, 1), 32, 0, stream>>>(xbuf, gsrc, gdst, gp.g0, f(B + 7), eH1);
    gemm<128, 8>(stream, eH1, gp.g1, f(B + 9), eH2, E, 128, 128, 1);
    k_scatter_max<<<cdiv((long long)E * 128, 256), 256, 0, stream>>>(eH2, gdst, xcat, 256, 128, E, 128);
    k_fixup_nonfinite<<<cdiv((long long)N * 256, 256), 256, 0, stream>>>(xcat, (long long)N * 256);
    // out MLP 256 -> 256
    gemm<256, 8>(stream, xcat, gp.o, f(B + 11), out, N, 256, 256, 1);
  };

  // x1 = GCU1(motion)
  gcu(GCU1, gp1, motion, 64, xA);
  // xg = segment_max(mlt2(x1), batch)
  gemm<256, 8>(stream, xA, Pm0, f(MLT2 + 1), g1, N, 512, 512, 1);
  gemm<512, 8>(stream, g1, Pm1, f(MLT2 + 3), g2, N, 1024, 1024, 1);
  launch_fill(xg, -__builtin_inff(), (long long)G * 1024, stream);
  k_segment_max<<<cdiv((long long)N * 1024, 256), 256, 0, stream>>>(g2, batch, xg, N, 1024);
  k_fixup_nonfinite<<<cdiv((long long)G * 1024, 256), 256, 0, stream>>>(xg, (long long)G * 1024);
  // x2 = GCU2(x1); x3 = GCU3(x2)
  gcu(GCU2, gp2, xA, 256, xB);
  gcu(GCU3, gp3, xB, 256, xA);
  // cls head
  k_gemm_cat<<<dim3(N / 16, 8), 32, 0, stream>>>(xA, xg, batch, Pc0, f(CLS + 1), g2);
  gemm<1024, 8>(stream, g2, Pc1, f(CLS + 3), g1, N, 512, 512, 1);
  // final 512 -> 5 (padded to M = 16), then extract 5 columns
  launch_fill(bfP, 0.f, 16, stream);
  k_copy_cols<<<1, 256, 0, stream>>>(bfP, 5, 0, f(FIN + 1), 5, 1);
  gemm<512, 1>(stream, g1, Pfin, bfP, Yf, N, 16, 16, 0);
  k_extract_cols<<<cdiv((long long)N * 5, 256), 256, 0, stream>>>((float*)d_out, 5, Yf, 16, 0, N);
}
